// DepthAdapterWindowAttn_39573828666160
// MI455X (gfx1250) — matrix operand order repaired and checked by reading, not compiled, not run on hardware
//
#include <hip/hip_runtime.h>
#include <hip/hip_bf16.h>

#define NTOK   131072          // 8 * 128 * 128
#define FEAT   106
#define KF     128             // padded feature K (4 x 32)
#define HID    384
#define CODE   90
#define PSTRIDE 112            // P column stride (7 tiles * 16)

typedef __attribute__((ext_vector_type(16))) __bf16 v16bf;
typedef __attribute__((ext_vector_type(8)))  float  v8f;

union BFrag { uint4 u[2]; v16bf v; };

__device__ __forceinline__ unsigned short f2bf(float x) {
  unsigned u = __float_as_uint(x);
  u += 0x7fffu + ((u >> 16) & 1u);           // round-to-nearest-even
  return (unsigned short)(u >> 16);
}

// ---------------------------------------------------------------------------
// Stage a contiguous 32-row x K-col bf16 tile (32*K*2 bytes) into LDS using
// the CDNA5 direct global->LDS async path (no VGPR staging, ASYNCcnt-tracked).
// Must be called by all 256 threads; caller must __syncthreads() after.
// ---------------------------------------------------------------------------
template<int K>
__device__ __forceinline__ void stage_tile_async(unsigned short* lds,
                                                 const unsigned short* gsrc) {
  constexpr int NV = 32 * K / 8;                 // # of 16-byte chunks
  unsigned base = (unsigned)(uintptr_t)lds;      // low 32 bits = LDS offset
  unsigned long long ga = (unsigned long long)(uintptr_t)gsrc;
  #pragma unroll
  for (int j = 0; j < NV / 256; ++j) {
    unsigned idx = threadIdx.x + j * 256;
    asm volatile("global_load_async_to_lds_b128 %0, %1, off"
                 :: "v"(base + 16u * idx), "v"(ga + 16ull * idx)
                 : "memory");
  }
  asm volatile("s_wait_asynccnt 0" ::: "memory");
}

// ---------------------------------------------------------------------------
// Pack an f32 [Kreal x Nreal] row-major weight into WMMA B-fragment layout:
// out[((kt*NT + nt)*32 + lane)*16 + e] = bf16(W[kt*32 + (e>>3)*16 + (lane>>4)*8 + (e&7)][nt*16 + lane&15])
// Zero-padded outside real dims so padded-K GEMMs are exact.
// ---------------------------------------------------------------------------
__global__ void pack_weight(const float* __restrict__ W, int Kreal, int Nreal,
                            int KT, int NT, unsigned short* __restrict__ out) {
  int idx = blockIdx.x * blockDim.x + threadIdx.x;
  int total = KT * NT * 512;
  if (idx >= total) return;
  int e    = idx & 15;
  int lane = (idx >> 4) & 31;
  int tile = idx >> 9;
  int nt = tile % NT, kt = tile / NT;
  int k = kt * 32 + (e >> 3) * 16 + (lane >> 4) * 8 + (e & 7);
  int n = nt * 16 + (lane & 15);
  float v = (k < Kreal && n < Nreal) ? W[(size_t)k * Nreal + n] : 0.f;
  out[idx] = f2bf(v);
}

// ---------------------------------------------------------------------------
// Build zero-padded bf16 feature matrix [NTOK x 128] and gate scalar g[NTOK].
// One wave per token; full-wave shfl reduction for the 106-d gate dot.
// ---------------------------------------------------------------------------
__global__ void __launch_bounds__(256) prep_feat(const float* __restrict__ codes,
                                                 const float* __restrict__ depth,
                                                 const float* __restrict__ gate_w,
                                                 const float* __restrict__ gate_b,
                                                 unsigned short* __restrict__ featbf,
                                                 float* __restrict__ g) {
  int t = blockIdx.x * 8 + (threadIdx.x >> 5);
  int lane = threadIdx.x & 31;
  if (t >= NTOK) return;
  float dot = 0.f;
  #pragma unroll
  for (int c = lane; c < KF; c += 32) {
    float v = 0.f;
    if (c < CODE)      v = codes[(size_t)t * CODE + c];
    else if (c < FEAT) v = depth[(size_t)t * 16 + (c - CODE)];
    featbf[(size_t)t * KF + c] = f2bf(v);
    if (c < FEAT) dot += v * gate_w[c];
  }
  #pragma unroll
  for (int off = 16; off; off >>= 1) dot += __shfl_xor(dot, off, 32);
  if (lane == 0) g[t] = dot + gate_b[0];
}

// ---------------------------------------------------------------------------
// Core: one wave computes S adjacent 32(M) x 16(N) f32 accumulator blocks.
// K-outer / N-inner: each A fragment is loaded from LDS once and reused for
// all S B-tiles (3x less ds traffic than N-outer).
// ---------------------------------------------------------------------------
template<int K, int S>
__device__ __forceinline__ void mma_tiles(const unsigned short* __restrict__ ldsX,
                                          const unsigned short* __restrict__ wp,
                                          int nt0, int NT, v8f (&acc)[S][2]) {
  const int lane = threadIdx.x & 31;
  const int m = lane & 15;
  const int h = lane >> 4;
  constexpr int KT = K / 32;
  #pragma unroll
  for (int kt = 0; kt < KT; ++kt) {
    BFrag a[2];
    #pragma unroll
    for (int mt = 0; mt < 2; ++mt) {
      const unsigned short* r = ldsX + (mt * 16 + m) * K + kt * 32 + 8 * h;
      a[mt].u[0] = *(const uint4*)r;          // K = 8h .. 8h+7
      a[mt].u[1] = *(const uint4*)(r + 16);   // K = 16+8h .. 23+8h
    }
    #pragma unroll
    for (int s = 0; s < S; ++s) {
      BFrag b;
      const uint4* bp =
          (const uint4*)(wp + (((size_t)kt * NT + nt0 + s) * 32 + lane) * 16);
      b.u[0] = bp[0];
      b.u[1] = bp[1];
      #pragma unroll
      for (int mt = 0; mt < 2; ++mt)
        acc[s][mt] = __builtin_amdgcn_wmma_f32_16x16x32_bf16(
            false, a[mt].v, false, b.v, (short)0, acc[s][mt], false, false);
    }
  }
}

// ---------------------------------------------------------------------------
// P = featbf @ Wp + bias   (f32 out, stride 112).  8 waves, waves 0..6 own
// one 16-col tile each over a 32-row M tile.
// ---------------------------------------------------------------------------
__global__ void __launch_bounds__(256) gemm_proj(const unsigned short* __restrict__ X,
                                                 const unsigned short* __restrict__ Wp,
                                                 const float* __restrict__ bias,
                                                 float* __restrict__ P) {
  __shared__ alignas(16) unsigned short ldsX[32 * KF];
  const int m0 = blockIdx.x * 32;
  stage_tile_async<KF>(ldsX, X + (size_t)m0 * KF);
  __syncthreads();
  const int wave = threadIdx.x >> 5;
  const int lane = threadIdx.x & 31;
  if (wave < 7) {
    v8f acc[1][2] = {};
    mma_tiles<KF, 1>(ldsX, Wp, wave, 7, acc);
    const int col = wave * 16 + (lane & 15);
    const float bv = (col < FEAT) ? bias[col] : 0.f;
    const int h = lane >> 4;
    #pragma unroll
    for (int mt = 0; mt < 2; ++mt)
      #pragma unroll
      for (int i = 0; i < 8; ++i) {
        int row = m0 + mt * 16 + i + 8 * h;
        P[(size_t)row * PSTRIDE + col] = acc[0][mt][i] + bv;
      }
  }
}

// ---------------------------------------------------------------------------
// 3x3 reflect-padded window soft attention.  2 tokens per 256-thread block;
// one thread per (token, channel).  Softmax weights + neighbor ids computed
// once per token in LDS.
// ---------------------------------------------------------------------------
__global__ void __launch_bounds__(256) attn(const float* __restrict__ P,
                                            const float* __restrict__ g,
                                            unsigned short* __restrict__ xbf) {
  __shared__ float wgt[2][9];
  __shared__ int   nb[2][9];
  const int local = threadIdx.x >> 7;    // 0..1
  const int c     = threadIdx.x & 127;
  const int t = blockIdx.x * 2 + local;
  if (c == 0) {
    const int b = t >> 14, r = t & 16383, i = r >> 7, j = r & 127;
    float lg[9]; float mx = -3.4e38f;
    #pragma unroll
    for (int di = -1; di <= 1; ++di)
      #pragma unroll
      for (int dj = -1; dj <= 1; ++dj) {
        int ri = i + di; ri = ri < 0 ? -ri : (ri > 127 ? 254 - ri : ri);
        int rj = j + dj; rj = rj < 0 ? -rj : (rj > 127 ? 254 - rj : rj);
        int k = (di + 1) * 3 + (dj + 1);
        int n = (b << 14) + (ri << 7) + rj;
        nb[local][k] = n;
        float v = g[n];
        lg[k] = v;
        mx = fmaxf(mx, v);
      }
    float s = 0.f;
    #pragma unroll
    for (int k = 0; k < 9; ++k) { lg[k] = __expf(lg[k] - mx); s += lg[k]; }
    float inv = 1.f / s;
    #pragma unroll
    for (int k = 0; k < 9; ++k) wgt[local][k] = lg[k] * inv;
  }
  __syncthreads();
  float acc = 0.f;
  if (c < FEAT) {
    #pragma unroll
    for (int k = 0; k < 9; ++k)
      acc += wgt[local][k] * P[(size_t)nb[local][k] * PSTRIDE + c];
  }
  xbf[(size_t)t * KF + c] = f2bf(acc);
}

// ---------------------------------------------------------------------------
// Y = relu(LN(X @ W + bias) * gamma + beta), bf16 out [NTOK x 384].
// 8 waves * 3 n-tiles each = 384 cols; f32 accumulators staged through an
// LDS tile for the cross-wave per-row LayerNorm.
// ---------------------------------------------------------------------------
template<int K>
__global__ void __launch_bounds__(256) gemm_ln(const unsigned short* __restrict__ X,
                                               const unsigned short* __restrict__ W,
                                               const float* __restrict__ bias,
                                               const float* __restrict__ gamma,
                                               const float* __restrict__ beta,
                                               unsigned short* __restrict__ Y) {
  __shared__ alignas(16) unsigned short ldsX[32 * K];
  __shared__ float ldsO[32 * HID];
  __shared__ float mu_s[32], rs_s[32];
  const int m0 = blockIdx.x * 32;
  stage_tile_async<K>(ldsX, X + (size_t)m0 * K);
  __syncthreads();
  const int wave = threadIdx.x >> 5;
  const int lane = threadIdx.x & 31;
  const int h = lane >> 4;
  v8f acc[3][2] = {};
  mma_tiles<K, 3>(ldsX, W, wave * 3, 24, acc);
  #pragma unroll
  for (int s = 0; s < 3; ++s) {
    const int col = (wave * 3 + s) * 16 + (lane & 15);
    #pragma unroll
    for (int mt = 0; mt < 2; ++mt)
      #pragma unroll
      for (int i = 0; i < 8; ++i)
        ldsO[(mt * 16 + i + 8 * h) * HID + col] = acc[s][mt][i];
  }
  __syncthreads();
  // per-row mean / var: 8 threads per row, shfl-xor reduce within the octet
  const int row = threadIdx.x >> 3;
  const int sub = threadIdx.x & 7;
  float sum = 0.f, sq = 0.f;
  for (int c = sub; c < HID; c += 8) {
    float v = ldsO[row * HID + c] + bias[c];
    sum += v; sq += v * v;
  }
  #pragma unroll
  for (int off = 1; off < 8; off <<= 1) {
    sum += __shfl_xor(sum, off, 32);
    sq  += __shfl_xor(sq,  off, 32);
  }
  if (sub == 0) {
    float mu = sum * (1.f / HID);
    float var = sq * (1.f / HID) - mu * mu;
    mu_s[row] = mu;
    rs_s[row] = rsqrtf(var + 1e-5f);
  }
  __syncthreads();
  for (int i = threadIdx.x; i < 32 * HID; i += 256) {
    int r = i / HID, c = i - r * HID;
    float v = ldsO[i] + bias[c];
    v = (v - mu_s[r]) * rs_s[r] * gamma[c] + beta[c];
    Y[(size_t)(m0 + r) * HID + c] = f2bf(fmaxf(v, 0.f));
  }
}

// ---------------------------------------------------------------------------
// out = codes + (h2 @ out_w + out_b), f32 [NTOK x 90].  Waves 0..5 own one
// 16-col tile (96 padded cols, only <90 stored).
// ---------------------------------------------------------------------------
__global__ void __launch_bounds__(256) gemm_out_k(const unsigned short* __restrict__ X,
                                                  const unsigned short* __restrict__ W,
                                                  const float* __restrict__ bias,
                                                  const float* __restrict__ codes,
                                                  float* __restrict__ out) {
  __shared__ alignas(16) unsigned short ldsX[32 * HID];
  const int m0 = blockIdx.x * 32;
  stage_tile_async<HID>(ldsX, X + (size_t)m0 * HID);
  __syncthreads();
  const int wave = threadIdx.x >> 5;
  const int lane = threadIdx.x & 31;
  if (wave < 6) {
    v8f acc[1][2] = {};
    mma_tiles<HID, 1>(ldsX, W, wave, 6, acc);
    const int col = wave * 16 + (lane & 15);
    const int h = lane >> 4;
    if (col < CODE) {
      const float bv = bias[col];
      #pragma unroll
      for (int mt = 0; mt < 2; ++mt)
        #pragma unroll
        for (int i = 0; i < 8; ++i) {
          int row = m0 + mt * 16 + i + 8 * h;
          out[(size_t)row * CODE + col] =
              codes[(size_t)row * CODE + col] + acc[0][mt][i] + bv;
        }
    }
  }
}

// ---------------------------------------------------------------------------
extern "C" void kernel_launch(void* const* d_in, const int* in_sizes, int n_in,
                              void* d_out, int out_size, void* d_ws, size_t ws_size,
                              hipStream_t stream) {
  (void)in_sizes; (void)n_in; (void)out_size; (void)ws_size;
  const float* codes = (const float*)d_in[0];
  const float* depth = (const float*)d_in[1];
  const float* apw   = (const float*)d_in[4];
  const float* apb   = (const float*)d_in[5];
  const float* agw   = (const float*)d_in[6];
  const float* agb   = (const float*)d_in[7];
  const float* w1    = (const float*)d_in[8];
  const float* b1    = (const float*)d_in[9];
  const float* ln1g  = (const float*)d_in[10];
  const float* ln1b  = (const float*)d_in[11];
  const float* w2    = (const float*)d_in[12];
  const float* b2    = (const float*)d_in[13];
  const float* ln2g  = (const float*)d_in[14];
  const float* ln2b  = (const float*)d_in[15];
  const float* ow    = (const float*)d_in[16];
  const float* ob    = (const float*)d_in[17];

  char* ws = (char*)d_ws;
  size_t off = 0;
  auto take = [&](size_t bytes) {
    char* p = ws + off;
    off = (off + bytes + 255) & ~(size_t)255;
    return p;
  };
  unsigned short* pWp    = (unsigned short*)take((size_t)4  * 7  * 512 * 2);
  unsigned short* pW1    = (unsigned short*)take((size_t)4  * 24 * 512 * 2);
  unsigned short* pW2    = (unsigned short*)take((size_t)12 * 24 * 512 * 2);
  unsigned short* pWo    = (unsigned short*)take((size_t)12 * 6  * 512 * 2);
  float*          g      = (float*)take((size_t)NTOK * 4);
  unsigned short* featbf = (unsigned short*)take((size_t)NTOK * KF * 2);
  float*          P      = (float*)take((size_t)NTOK * PSTRIDE * 4);
  unsigned short* xbf    = (unsigned short*)take((size_t)NTOK * KF * 2);
  unsigned short* h1     = (unsigned short*)take((size_t)NTOK * HID * 2);
  unsigned short* h2     = (unsigned short*)take((size_t)NTOK * HID * 2);

  pack_weight<<<(4 * 7 * 512 + 255) / 256, 256, 0, stream>>>(apw, FEAT, FEAT, 4, 7, pWp);
  pack_weight<<<(4 * 24 * 512 + 255) / 256, 256, 0, stream>>>(w1, FEAT, HID, 4, 24, pW1);
  pack_weight<<<(12 * 24 * 512 + 255) / 256, 256, 0, stream>>>(w2, HID, HID, 12, 24, pW2);
  pack_weight<<<(12 * 6 * 512 + 255) / 256, 256, 0, stream>>>(ow, HID, CODE, 12, 6, pWo);

  prep_feat<<<NTOK / 8, 256, 0, stream>>>(codes, depth, agw, agb, featbf, g);
  gemm_proj<<<NTOK / 32, 256, 0, stream>>>(featbf, pWp, apb, P);
  attn<<<NTOK / 2, 256, 0, stream>>>(P, g, xbf);
  gemm_ln<KF><<<NTOK / 32, 256, 0, stream>>>(xbf, pW1, b1, ln1g, ln1b, h1);
  gemm_ln<HID><<<NTOK / 32, 256, 0, stream>>>(h1, pW2, b2, ln2g, ln2b, h2);
  gemm_out_k<<<NTOK / 32, 256, 0, stream>>>(h2, pWo, ob, codes, (float*)d_out);
}
